// GAT_20641612825051
// MI455X (gfx1250) — matrix operand order repaired and checked by reading, not compiled, not run on hardware
//
#include <hip/hip_runtime.h>
#include <hip/hip_bf16.h>

typedef __attribute__((ext_vector_type(16))) _Float16 v16h;
typedef __attribute__((ext_vector_type(8)))  float    v8f;

#define NEG_SLOPE 0.2f

// ---------------------------------------------------------------------------
// Pre-pack W[K,NC] (f32, row-major) into B-fragment layout, f16, zero-padded:
//   P[((ct*nkb + kbi)*32 + lane)*16 + i] = f16(W[(kbi*32 + (lane>>4)*16 + i)*NC
//                                               + ct*16 + (lane&15)])
// One thread per (fragment, lane); each writes a contiguous 32B v16h.
// ---------------------------------------------------------------------------
__global__ void gat_pack_w(const float* __restrict__ W, _Float16* __restrict__ P,
                           int K, int NC, int NCt) {
    const int idx = blockIdx.x * blockDim.x + threadIdx.x;
    const int nkb = K >> 5;
    if (idx >= NCt * nkb * 32) return;
    const int lane = idx & 31;
    const int f    = idx >> 5;
    const int kbi  = f % nkb;
    const int ct   = f / nkb;
    const int col  = ct * 16 + (lane & 15);
    const int k0   = (kbi << 5) + (lane >> 4) * 16;
    v16h b;
#pragma unroll
    for (int i = 0; i < 16; ++i) {
        float wv = (col < NC) ? W[(size_t)(k0 + i) * NC + col] : 0.0f;
        b[i] = (_Float16)wv;
    }
    *(v16h*)(P + (size_t)idx * 16) = b;
}

// ---------------------------------------------------------------------------
// WMMA GEMM: Hout[N,NC] = A[N,K] * W[K,NC]; A f32 converted on the fly, B from
// pre-packed f16 fragments. One wave per 16x16 tile; wave w = column tile w.
// N % 16 == 0, K % 32 == 0. Inner loop is branch-free, all lanes active.
// ---------------------------------------------------------------------------
__global__ void gat_gemm_wmma(const float* __restrict__ A,
                              const _Float16* __restrict__ P,
                              float* __restrict__ Hout,
                              int N, int K, int NC) {
    const int lane = threadIdx.x & 31;
    const int wave = threadIdx.x >> 5;
    const int nkb  = K >> 5;
    const int rbase = blockIdx.x * 16;
    const int l  = lane & 15;
    const int hi = lane >> 4;
    const int row = rbase + l;
    const int col = wave * 16 + l;

    // A fragment source: lanes 0-15 K=kb+0..15; lanes 16-31 K=kb+16..31.
    const float*    ap = A + (size_t)row * K + hi * 16;
    const _Float16* bp = P + ((size_t)wave * nkb * 32 + lane) * 16;

    v8f acc = {};
    for (int kbi = 0; kbi < nkb; ++kbi) {
        v16h a;
#pragma unroll
        for (int i = 0; i < 8; ++i) {
            a[i]     = (_Float16)ap[i];
            a[i + 8] = (_Float16)ap[i + 8];
        }
        v16h b = *(const v16h*)bp;
        acc = __builtin_amdgcn_wmma_f32_16x16x32_f16(
            /*neg_a=*/false, a, /*neg_b=*/false, b,
            /*c_mod=*/(short)0, acc, /*reuse_a=*/false, /*reuse_b=*/false);
        ap += 32;
        bp += 32 * 16;
    }
    if (col < NC) {
        const int row0 = rbase + hi * 8;   // C/D: VGPR v -> row rbase+hi*8+v
#pragma unroll
        for (int v = 0; v < 8; ++v)
            Hout[(size_t)(row0 + v) * NC + col] = acc[v];
    }
}

// el[n,h] = dot(h[n,h,:], al[h,:]) ; er likewise. One thread per (n,h).
__global__ void gat_scores(const float* __restrict__ Hf,
                           const float* __restrict__ al,
                           const float* __restrict__ ar,
                           float* __restrict__ el, float* __restrict__ er,
                           int N, int H, int D) {
    int i = blockIdx.x * blockDim.x + threadIdx.x;
    if (i >= N * H) return;
    int n = i / H, h = i % H;
    const float* hp  = Hf + (size_t)n * H * D + (size_t)h * D;
    const float* alp = al + h * D;
    const float* arp = ar + h * D;
    float sl = 0.f, sr = 0.f;
    for (int d = 0; d < D; ++d) {
        float v = hp[d];
        sl += v * alp[d];
        sr += v * arp[d];
    }
    el[i] = sl;
    er[i] = sr;
}

__global__ void gat_fill(float* __restrict__ p, float val, int n) {
    int i = blockIdx.x * blockDim.x + threadIdx.x;
    if (i < n) p[i] = val;
}

__global__ void gat_init_bias(float* __restrict__ out,
                              const float* __restrict__ bias, int N, int F) {
    int i = blockIdx.x * blockDim.x + threadIdx.x;
    if (i < N * F) out[i] = bias[i % F];
}

__device__ __forceinline__ void atomicMaxF32(float* addr, float val) {
    if (val >= 0.0f)
        atomicMax((int*)addr, __float_as_int(val));
    else
        atomicMin((unsigned int*)addr, __float_as_uint(val));
}

__device__ __forceinline__ float leaky(float x) {
    return x > 0.0f ? x : NEG_SLOPE * x;
}

// --- H==4 specializations: one float4 load per endpoint ---------------------
__global__ void gat_edge_max4(const int* __restrict__ src, const int* __restrict__ dst,
                              const float4* __restrict__ el4, const float4* __restrict__ er4,
                              float* __restrict__ m, int E) {
    int e = blockIdx.x * blockDim.x + threadIdx.x;
    if (e >= E) return;
    int s = src[e], d = dst[e];
    float4 a = el4[s], b = er4[d];
    atomicMaxF32(&m[d * 4 + 0], leaky(a.x + b.x));
    atomicMaxF32(&m[d * 4 + 1], leaky(a.y + b.y));
    atomicMaxF32(&m[d * 4 + 2], leaky(a.z + b.z));
    atomicMaxF32(&m[d * 4 + 3], leaky(a.w + b.w));
}

__global__ void gat_edge_sum4(const int* __restrict__ src, const int* __restrict__ dst,
                              const float4* __restrict__ el4, const float4* __restrict__ er4,
                              const float4* __restrict__ m4, float* __restrict__ den, int E) {
    int e = blockIdx.x * blockDim.x + threadIdx.x;
    if (e >= E) return;
    int s = src[e], d = dst[e];
    float4 a = el4[s], b = er4[d], mm = m4[d];
    atomicAdd(&den[d * 4 + 0], expf(leaky(a.x + b.x) - mm.x));
    atomicAdd(&den[d * 4 + 1], expf(leaky(a.y + b.y) - mm.y));
    atomicAdd(&den[d * 4 + 2], expf(leaky(a.z + b.z) - mm.z));
    atomicAdd(&den[d * 4 + 3], expf(leaky(a.w + b.w) - mm.w));
}

// --- generic (used for H==1) ------------------------------------------------
__global__ void gat_edge_max(const int* __restrict__ src, const int* __restrict__ dst,
                             const float* __restrict__ el, const float* __restrict__ er,
                             float* __restrict__ m, int E, int H) {
    int e = blockIdx.x * blockDim.x + threadIdx.x;
    if (e >= E) return;
    int s = src[e], d = dst[e];
    for (int h = 0; h < H; ++h)
        atomicMaxF32(&m[d * H + h], leaky(el[s * H + h] + er[d * H + h]));
}

__global__ void gat_edge_sum(const int* __restrict__ src, const int* __restrict__ dst,
                             const float* __restrict__ el, const float* __restrict__ er,
                             const float* __restrict__ m, float* __restrict__ den,
                             int E, int H) {
    int e = blockIdx.x * blockDim.x + threadIdx.x;
    if (e >= E) return;
    int s = src[e], d = dst[e];
    for (int h = 0; h < H; ++h) {
        float v = leaky(el[s * H + h] + er[d * H + h]);
        atomicAdd(&den[d * H + h], expf(v - m[d * H + h]));
    }
}

// Pass 3: out[dst] += alpha * h[src]. One block per edge; alpha computed by
// the first H lanes, shared via LDS, scattered by F lanes with f32 atomics.
__global__ void gat_edge_aggr(const int* __restrict__ src, const int* __restrict__ dst,
                              const float* __restrict__ el, const float* __restrict__ er,
                              const float* __restrict__ m, const float* __restrict__ den,
                              const float* __restrict__ Hf, float* __restrict__ out,
                              int H, int D) {
    __shared__ float s_alpha[4];
    __shared__ int s_sd[2];
    const int e = blockIdx.x;
    const int t = threadIdx.x;
    if (t == 0) { s_sd[0] = src[e]; s_sd[1] = dst[e]; }
    __syncthreads();
    const int s = s_sd[0], d = s_sd[1];
    if (t < H) {
        float v  = leaky(el[s * H + t] + er[d * H + t]);
        float ex = expf(v - m[d * H + t]);
        s_alpha[t] = ex / (den[d * H + t] + 1e-9f);
    }
    __syncthreads();
    const int F = H * D;
    if (t < F) {
        atomicAdd(&out[(size_t)d * F + t], s_alpha[t / D] * Hf[(size_t)s * F + t]);
    }
}

__global__ void gat_elu(float* __restrict__ p, int n) {
    int i = blockIdx.x * blockDim.x + threadIdx.x;
    if (i < n) {
        float v = p[i];
        p[i] = v > 0.0f ? v : (expf(v) - 1.0f);
    }
}

// ---------------------------------------------------------------------------
// One GAT layer.
// ---------------------------------------------------------------------------
static void run_layer(const float* feat, int K,
                      const int* src, const int* dst,
                      const float* W, const float* al, const float* ar,
                      const float* bias, int H, int D,
                      float* hbuf, float* el, float* er, float* m, float* den,
                      _Float16* wpack, float* out, int N, int E,
                      hipStream_t stream) {
    const int F   = H * D;
    const int NCt = (F + 15) / 16;
    const int nkb = K >> 5;

    const int npack = NCt * nkb * 32;
    gat_pack_w<<<(npack + 255) / 256, 256, 0, stream>>>(W, wpack, K, F, NCt);
    gat_gemm_wmma<<<N / 16, NCt * 32, 0, stream>>>(feat, wpack, hbuf, N, K, F);

    const int nh = N * H;
    gat_scores<<<(nh + 255) / 256, 256, 0, stream>>>(hbuf, al, ar, el, er, N, H, D);
    gat_fill<<<(nh + 255) / 256, 256, 0, stream>>>(m, -3.402823466e+38f, nh);
    gat_fill<<<(nh + 255) / 256, 256, 0, stream>>>(den, 0.0f, nh);

    const int eb = (E + 255) / 256;
    if (H == 4) {
        gat_edge_max4<<<eb, 256, 0, stream>>>(src, dst, (const float4*)el,
                                              (const float4*)er, m, E);
        gat_edge_sum4<<<eb, 256, 0, stream>>>(src, dst, (const float4*)el,
                                              (const float4*)er, (const float4*)m,
                                              den, E);
    } else {
        gat_edge_max<<<eb, 256, 0, stream>>>(src, dst, el, er, m, E, H);
        gat_edge_sum<<<eb, 256, 0, stream>>>(src, dst, el, er, m, den, E, H);
    }

    const int nf = N * F;
    gat_init_bias<<<(nf + 255) / 256, 256, 0, stream>>>(out, bias, N, F);

    const int bt = ((F + 63) / 64) * 64;           // 128 for F=128, 64 for F=40
    gat_edge_aggr<<<E, bt, 0, stream>>>(src, dst, el, er, m, den, hbuf, out, H, D);
}

extern "C" void kernel_launch(void* const* d_in, const int* in_sizes, int n_in,
                              void* d_out, int out_size, void* d_ws, size_t ws_size,
                              hipStream_t stream) {
    (void)n_in; (void)out_size; (void)ws_size;
    const float* x    = (const float*)d_in[0];
    const int*   src  = (const int*)  d_in[1];
    const int*   dst  = (const int*)  d_in[2];
    const float* W1   = (const float*)d_in[3];
    const float* al1  = (const float*)d_in[4];
    const float* ar1  = (const float*)d_in[5];
    const float* b1   = (const float*)d_in[6];
    const float* W2   = (const float*)d_in[7];
    const float* al2  = (const float*)d_in[8];
    const float* ar2  = (const float*)d_in[9];
    const float* b2   = (const float*)d_in[10];
    const float* W3   = (const float*)d_in[11];
    const float* al3  = (const float*)d_in[12];
    const float* ar3  = (const float*)d_in[13];
    const float* b3   = (const float*)d_in[14];

    const int FIN = 256, H = 4, D = 32, F1 = 128, NC = 40;
    const int N = in_sizes[0] / FIN;     // 100000
    const int E = in_sizes[1];           // 1600000

    // Workspace (floats): [hbuf N*128][outb N*128][el N*4][er N*4][m N*4][den N*4]
    // then a 64KB f16 packed-W buffer (32B aligned by construction).
    float* hbuf = (float*)d_ws;
    float* outb = hbuf + (size_t)N * F1;
    float* el   = outb + (size_t)N * F1;
    float* er   = el   + (size_t)N * H;
    float* m    = er   + (size_t)N * H;
    float* den  = m    + (size_t)N * H;
    _Float16* wpack = (_Float16*)(den + (size_t)N * H);

    // Layer 1: x[N,256] -> outb[N,128], ELU in place.
    run_layer(x, FIN, src, dst, W1, al1, ar1, b1, H, D,
              hbuf, el, er, m, den, wpack, outb, N, E, stream);
    gat_elu<<<((size_t)N * F1 + 255) / 256, 256, 0, stream>>>(outb, N * F1);

    // Layer 2: outb[N,128] -> outb[N,128] (GEMM consumes input before re-init).
    run_layer(outb, F1, src, dst, W2, al2, ar2, b2, H, D,
              hbuf, el, er, m, den, wpack, outb, N, E, stream);
    gat_elu<<<((size_t)N * F1 + 255) / 256, 256, 0, stream>>>(outb, N * F1);

    // Layer 3: outb[N,128] -> d_out[N,40] (single head, no activation).
    run_layer(outb, F1, src, dst, W3, al3, ar3, b3, 1, NC,
              hbuf, el, er, m, den, wpack, (float*)d_out, N, E, stream);
}
